// Non_local_47923245088807
// MI455X (gfx1250) — matrix operand order repaired and checked by reading, not compiled, not run on hardware
//
#include <hip/hip_runtime.h>
#include <hip/hip_bf16.h>
#include <stdint.h>

// ---------------------------------------------------------------------------
// Non-local block, algebraically refactored (no softmax => attention is a
// linear map):
//   S  = phi_proj @ g_proj^T-ish     [128x128]  per batch  (K = N = 2048)
//   M  = (W_w @ S^T) / N             [256x128]  per batch
//   z  = M @ theta_proj + W_b        [256x2048] per batch
//   out = BN(z) + x
// All heavy GEMMs use v_wmma_f32_16x16x32_bf16 (bf16 in, f32 accum).
// Problem is memory-bound (~150 MB traffic @ 23.3 TB/s ~ 6.5 us floor).
// ---------------------------------------------------------------------------

typedef __attribute__((ext_vector_type(16))) __bf16 v16bf;
typedef __attribute__((ext_vector_type(2)))  __bf16 bf16x2;
typedef __attribute__((ext_vector_type(8)))  float  v8f;
typedef __attribute__((ext_vector_type(4)))  float  f32x4;

#define LDT 34   // padded LDS row stride (32 bf16 + 2 pad) -> conflict-free, 4B aligned

__device__ __forceinline__ bf16x2 ld_bf16x2(const __bf16* p) {
  uint32_t u = *reinterpret_cast<const uint32_t*>(p);
  return __builtin_bit_cast(bf16x2, u);
}

// A-fragment: 16x32 bf16, M=row, ISA layout: lanes 0-15 hold K{0..7,16..23},
// lanes 16-31 hold K{8..15,24..31}; element e -> k = kb + (e<8 ? e : 8+e).
__device__ __forceinline__ v16bf load_fragA(const __bf16* As, int rowtile, int lane) {
  const int m  = rowtile * 16 + (lane & 15);
  const int kb = (lane & 16) ? 8 : 0;
  const __bf16* row = As + m * LDT;
  v16bf f;
#pragma unroll
  for (int j = 0; j < 4; ++j) {
    bf16x2 lo = ld_bf16x2(row + kb + 2 * j);
    bf16x2 hi = ld_bf16x2(row + kb + 16 + 2 * j);
    f[2 * j]     = lo[0]; f[2 * j + 1] = lo[1];
    f[8 + 2 * j] = hi[0]; f[9 + 2 * j] = hi[1];
  }
  return f;
}

// B-fragment: 32x16 bf16, N=col; lanes 0-15 hold K{0..7,16..23}, lanes 16-31 K{8..15,24..31}.
// Bs is staged transposed as [n][k] so pairs are contiguous.
__device__ __forceinline__ v16bf load_fragB(const __bf16* Bs, int coltile, int lane) {
  const int n  = coltile * 16 + (lane & 15);
  const int kb = (lane & 16) ? 8 : 0;
  const __bf16* row = Bs + n * LDT + kb;
  v16bf f;
#pragma unroll
  for (int j = 0; j < 8; ++j) {
    bf16x2 d = ld_bf16x2(row + ((j & 4) << 2) + 2 * (j & 3));
    f[2 * j] = d[0]; f[2 * j + 1] = d[1];
  }
  return f;
}

// Generic batched GEMM:  C[b] = scale * (A[b] @ B[b]) + bias(row)
// A: [M,K] f32 row-major (batch stride sA; 0 = shared weights)
// B: [K,N] f32 row-major (batch stride sB)
// C: f32; outT==0 -> [M,N] row-major, outT==1 -> stored transposed [N,M]
// Requires M,N multiples of 128 and K multiple of 32 (true for all calls).
__global__ __launch_bounds__(256)
void gemm_bf16_wmma(const float* __restrict__ A, const float* __restrict__ B,
                    float* __restrict__ C, const float* __restrict__ bias,
                    float scale, int M, int N, int K,
                    long long sA, long long sB, long long sC, int outT)
{
  __shared__ __bf16 As[128 * LDT];
  __shared__ __bf16 Bs[128 * LDT];

  const int bz = blockIdx.z;
  const float* Ab = A + (long long)bz * sA;
  const float* Bb = B + (long long)bz * sB;
  float*       Cb = C + (long long)bz * sC;

  const int row0 = blockIdx.y * 128;
  const int col0 = blockIdx.x * 128;
  const int tid  = threadIdx.x;
  const int lane = tid & 31;
  const int wid  = tid >> 5;
  const int wrow = wid >> 2;  // 0..1 -> 64 rows each
  const int wcol = wid & 3;   // 0..3 -> 32 cols each

  const v8f zero8 = {0.f, 0.f, 0.f, 0.f, 0.f, 0.f, 0.f, 0.f};
  v8f acc[4][2];
#pragma unroll
  for (int r = 0; r < 4; ++r)
#pragma unroll
    for (int c = 0; c < 2; ++c) acc[r][c] = zero8;

  for (int k0 = 0; k0 < K; k0 += 32) {
    // ---- stage A tile: 128 rows x 32 k  (f32 -> bf16) ----
#pragma unroll
    for (int i = 0; i < 4; ++i) {
      int lin = i * 256 + tid;          // 0..1023 float4 slots
      int r   = lin >> 3;               // row 0..127
      int kv  = (lin & 7) << 2;         // k offset 0..28
      f32x4 v = *(const f32x4*)(Ab + (long long)(row0 + r) * K + k0 + kv);
      __bf16* dst = As + r * LDT + kv;
      dst[0] = (__bf16)v.x; dst[1] = (__bf16)v.y;
      dst[2] = (__bf16)v.z; dst[3] = (__bf16)v.w;
    }
    // ---- stage B tile: 32 k-rows x 128 cols, transposed into Bs[n][k] ----
#pragma unroll
    for (int i = 0; i < 4; ++i) {
      int lin = i * 256 + tid;
      int r   = lin >> 5;               // k row 0..31
      int c4  = (lin & 31) << 2;        // col 0..124
      f32x4 v = *(const f32x4*)(Bb + (long long)(k0 + r) * N + col0 + c4);
      Bs[(c4 + 0) * LDT + r] = (__bf16)v.x;
      Bs[(c4 + 1) * LDT + r] = (__bf16)v.y;
      Bs[(c4 + 2) * LDT + r] = (__bf16)v.z;
      Bs[(c4 + 3) * LDT + r] = (__bf16)v.w;
    }
    // prefetch next K tile into cache hierarchy (global_prefetch_b8)
    if (k0 + 32 < K) {
      __builtin_prefetch(Ab + (long long)(row0 + (tid >> 1)) * K + k0 + 32 + ((tid & 1) << 4), 0, 1);
      __builtin_prefetch(Bb + (long long)(k0 + 32 + (tid >> 3)) * N + col0 + ((tid & 7) << 4), 0, 1);
    }
    __syncthreads();

    v16bf bfrag[2];
#pragma unroll
    for (int ct = 0; ct < 2; ++ct) bfrag[ct] = load_fragB(Bs, wcol * 2 + ct, lane);
#pragma unroll
    for (int rt = 0; rt < 4; ++rt) {
      v16bf afrag = load_fragA(As, wrow * 4 + rt, lane);
#pragma unroll
      for (int ct = 0; ct < 2; ++ct) {
        acc[rt][ct] = __builtin_amdgcn_wmma_f32_16x16x32_bf16(
            false, afrag, false, bfrag[ct], (short)0, acc[rt][ct], false, false);
      }
    }
    __syncthreads();
  }

  // ---- epilogue: scale + bias, optional transposed store ----
  const int moff = (lane >> 4) << 3;  // +8 rows for lanes 16..31
  const int nl   = lane & 15;
#pragma unroll
  for (int rt = 0; rt < 4; ++rt) {
#pragma unroll
    for (int ct = 0; ct < 2; ++ct) {
#pragma unroll
      for (int v = 0; v < 8; ++v) {
        int gm = row0 + wrow * 64 + rt * 16 + v + moff;
        int gn = col0 + wcol * 32 + ct * 16 + nl;
        float val = acc[rt][ct][v] * scale;
        if (bias) val += bias[gm];
        if (outT) Cb[(long long)gn * M + gm] = val;
        else      Cb[(long long)gm * N + gn] = val;
      }
    }
  }
}

// Deterministic per-channel batch stats over (B, N): mean and 1/sqrt(var+eps)
__global__ __launch_bounds__(256)
void bn_stats(const float* __restrict__ z, float* __restrict__ meanv,
              float* __restrict__ invstd, int Bn, int Cn, int Nn)
{
  const int c = blockIdx.x;
  const int tid = threadIdx.x;
  float s = 0.f, s2 = 0.f;
  for (int b = 0; b < Bn; ++b) {
    const float* p = z + ((long long)b * Cn + c) * Nn;
    for (int n = tid; n < Nn; n += 256) { float v = p[n]; s += v; s2 += v * v; }
  }
  __shared__ float sh[256], sh2[256];
  sh[tid] = s; sh2[tid] = s2;
  __syncthreads();
  for (int off = 128; off; off >>= 1) {
    if (tid < off) { sh[tid] += sh[tid + off]; sh2[tid] += sh2[tid + off]; }
    __syncthreads();
  }
  if (tid == 0) {
    float cnt = (float)Bn * (float)Nn;
    float m   = sh[0] / cnt;
    float var = sh2[0] / cnt - m * m;
    meanv[c]  = m;
    invstd[c] = rsqrtf(var + 1e-5f);
  }
}

// out = (z - mean)*invstd*gamma + beta + x
__global__ __launch_bounds__(256)
void bn_apply(const float* __restrict__ z, const float* __restrict__ x,
              const float* __restrict__ meanv, const float* __restrict__ invstd,
              const float* __restrict__ gamma, const float* __restrict__ beta,
              float* __restrict__ out, int Cn, int Nn)
{
  long long i = (long long)blockIdx.x * 256 + threadIdx.x;
  int c = (int)((i / Nn) % Cn);
  out[i] = (z[i] - meanv[c]) * invstd[c] * gamma[c] + beta[c] + x[i];
}

extern "C" void kernel_launch(void* const* d_in, const int* in_sizes, int n_in,
                              void* d_out, int out_size, void* d_ws, size_t ws_size,
                              hipStream_t stream)
{
  (void)in_sizes; (void)n_in; (void)out_size; (void)ws_size;
  const float* x       = (const float*)d_in[0];
  const float* g_w     = (const float*)d_in[1];
  const float* g_b     = (const float*)d_in[2];
  const float* theta_w = (const float*)d_in[3];
  const float* theta_b = (const float*)d_in[4];
  const float* phi_w   = (const float*)d_in[5];
  const float* phi_b   = (const float*)d_in[6];
  const float* W_w     = (const float*)d_in[7];
  const float* W_b     = (const float*)d_in[8];
  const float* gamma   = (const float*)d_in[9];
  const float* beta    = (const float*)d_in[10];
  float* out = (float*)d_out;

  const int B = 16, C = 256, I = 128, N = 2048;  // N = H*W

  // workspace layout (floats); total ~21.8M floats (~87 MB)
  float* ws = (float*)d_ws;
  float* proj_theta = ws;                          // [B][128][2048]
  float* proj_phi   = proj_theta + (size_t)B * I * N;
  float* g_token    = proj_phi   + (size_t)B * I * N;   // [B][2048][128]
  float* S_T        = g_token    + (size_t)B * I * N;   // [B][128][128] = S^T
  float* Mmat       = S_T        + (size_t)B * I * I;   // [B][256][128]
  float* zbuf       = Mmat       + (size_t)B * C * I;   // [B][256][2048]
  float* meanv      = zbuf       + (size_t)B * C * N;   // [256]
  float* invstd     = meanv      + C;                   // [256]

  dim3 blk(256);

  // 1) projections: W_p[128x256] @ x[b][256x2048] + b_p
  gemm_bf16_wmma<<<dim3(N / 128, 1, B), blk, 0, stream>>>(
      theta_w, x, proj_theta, theta_b, 1.f, I, N, C, 0, (long long)C * N, (long long)I * N, 0);
  gemm_bf16_wmma<<<dim3(N / 128, 1, B), blk, 0, stream>>>(
      phi_w, x, proj_phi, phi_b, 1.f, I, N, C, 0, (long long)C * N, (long long)I * N, 0);
  gemm_bf16_wmma<<<dim3(N / 128, 1, B), blk, 0, stream>>>(
      g_w, x, g_token, g_b, 1.f, I, N, C, 0, (long long)C * N, (long long)I * N, 1 /*token-major*/);

  // 2) S^T[b] : S[d,e] = sum_m phi[d,m] * g_token[m,e]; stored transposed
  gemm_bf16_wmma<<<dim3(1, 1, B), blk, 0, stream>>>(
      proj_phi, g_token, S_T, nullptr, 1.f, I, I, N,
      (long long)I * N, (long long)I * N, (long long)I * I, 1);

  // 3) M[b] = (1/N) * W_w[256x128] @ S_T[b][128x128]
  gemm_bf16_wmma<<<dim3(1, C / 128, B), blk, 0, stream>>>(
      W_w, S_T, Mmat, nullptr, 1.f / (float)N, C, I, I,
      0, (long long)I * I, (long long)C * I, 0);

  // 4) z[b] = M[b][256x128] @ proj_theta[b][128x2048] + W_b
  gemm_bf16_wmma<<<dim3(N / 128, C / 128, B), blk, 0, stream>>>(
      Mmat, proj_theta, zbuf, W_b, 1.f, C, N, I,
      (long long)C * I, (long long)I * N, (long long)C * N, 0);

  // 5) batch-norm statistics (training mode, population var), deterministic
  bn_stats<<<dim3(C), blk, 0, stream>>>(zbuf, meanv, invstd, B, C, N);

  // 6) normalize + affine + residual
  long long total = (long long)B * C * N;
  bn_apply<<<dim3((unsigned)(total / 256)), blk, 0, stream>>>(
      zbuf, x, meanv, invstd, gamma, beta, out, C, N);
}
